// MultiHeadAttention_49684181680249
// MI455X (gfx1250) — compile-verified, matrix operand order checked by reading, not run on hardware
//
#include <hip/hip_runtime.h>

#define EMB   768
#define HEADS 4
#define DH    192      // EMB / HEADS
#define BATCH 8
#define SEQ   2048

typedef __attribute__((ext_vector_type(16))) __bf16 v16bf;
typedef __attribute__((ext_vector_type(8)))  float  v8f;

// One 16x32 bf16 WMMA operand per lane: 16 halves = 32 bytes = 8 VGPRs.
// Per-lane element e maps to (row = lane%16, K = (e/8)*16 + (lane/16)*8 + e%8).
union Tile {
    v16bf bf;
    uint4 q[2];
    unsigned short s[16];
};

// Native bf16 convert (gfx1250 has bf16 datapaths).
__device__ __forceinline__ unsigned short f2bf(float f) {
    __bf16 h = (__bf16)f;
    return __builtin_bit_cast(unsigned short, h);
}

__device__ __forceinline__ v8f wmma_bf16(const Tile& a, const Tile& b, v8f c) {
    // D(16x16,f32) = A(16x32,bf16) x B(32x16,bf16) + C
    return __builtin_amdgcn_wmma_f32_16x16x32_bf16(
        false, a.bf, false, b.bf, (short)0, c, false, false);
}

// Load 16 bf16 (two 8-wide contiguous runs at p and p+16) as two 16B loads.
// Works for both global and (address-space-inferred) LDS pointers.
__device__ __forceinline__ Tile load_bf16_tile(const unsigned short* __restrict__ p) {
    Tile t;
    t.q[0] = *(const uint4*)(p);
    t.q[1] = *(const uint4*)(p + 16);
    return t;
}

// Async global->LDS copy of 16B per lane (CDNA5, ASYNCcnt-tracked, no VGPRs).
__device__ __forceinline__ void async_copy16(unsigned lds_off, const void* gaddr) {
    asm volatile("global_load_async_to_lds_b128 %0, %1, off"
                 :: "v"(lds_off), "v"((unsigned long long)gaddr) : "memory");
}
__device__ __forceinline__ void wait_async() {
    asm volatile("s_wait_asynccnt 0" ::: "memory");
}

// ---------------------------------------------------------------------------
// Elementwise fp32 -> bf16 (row-major copy). Hoists ALL conversion VALU out
// of the GEMM hot loops. One float4 -> ushort4 per thread.
// ---------------------------------------------------------------------------
__global__ __launch_bounds__(256)
void cvt_bf16_kernel(const float* __restrict__ in, unsigned short* __restrict__ out) {
    const size_t i = ((size_t)blockIdx.x * blockDim.x + threadIdx.x) * 4;
    float4 f = *(const float4*)(in + i);
    ushort4 o;
    o.x = f2bf(f.x); o.y = f2bf(f.y); o.z = f2bf(f.z); o.w = f2bf(f.w);
    *(ushort4*)(out + i) = o;
}

// ---------------------------------------------------------------------------
// Projection GEMM: out = X(16384x768) @ W^T(768x768) + bias, all bf16 inputs.
// Each wave computes a 32x64 tile: 8 WMMA per 32-wide K-chunk, 6 tile loads.
// transposed==0: out[((b*H+h)*S+s)*DH + dh]   (Q, K)
// transposed==1: out[((b*H+h)*DH+dh)*S + s]   (V, pre-transposed for P@V)
// ---------------------------------------------------------------------------
__global__ __launch_bounds__(256)
void proj_kernel(const unsigned short* __restrict__ Xb,
                 const unsigned short* __restrict__ Wb,
                 const float* __restrict__ bias, unsigned short* __restrict__ out,
                 int transposed)
{
    const int wave = (blockIdx.x * blockDim.x + threadIdx.x) >> 5;
    const int lane = threadIdx.x & 31;
    const int r = lane & 15, hi = lane >> 4;
    const int NT64 = EMB / 64;                 // 12
    const int tn64 = wave % NT64;
    const int tm   = wave / NT64;              // 0..511 (32-row tiles)
    const int row0 = tm * 32;
    const int n0   = tn64 * 64;

    v8f acc[2][4] = {};
    for (int k0 = 0; k0 < EMB; k0 += 32) {
        Tile a0 = load_bf16_tile(Xb + (size_t)(row0 +      r) * EMB + k0 + hi * 8);
        Tile a1 = load_bf16_tile(Xb + (size_t)(row0 + 16 + r) * EMB + k0 + hi * 8);
        #pragma unroll
        for (int t = 0; t < 4; ++t) {
            Tile b = load_bf16_tile(Wb + (size_t)(n0 + t * 16 + r) * EMB + k0 + hi * 8);
            acc[0][t] = wmma_bf16(a0, b, acc[0][t]);
            acc[1][t] = wmma_bf16(a1, b, acc[1][t]);
        }
    }

    #pragma unroll
    for (int ms = 0; ms < 2; ++ms) {
        const int rb     = row0 + ms * 16;
        const int bb     = rb / SEQ;
        const int s_base = rb % SEQ;
        #pragma unroll
        for (int t = 0; t < 4; ++t) {
            const int n  = n0 + t * 16 + r;
            const int h  = n / DH, dh = n % DH;
            const float bval = bias[n];
            #pragma unroll
            for (int e = 0; e < 8; ++e) {
                const int ss = s_base + e + 8 * hi;
                const float v = acc[ms][t][e] + bval;
                const size_t idx = transposed
                    ? ((size_t)((bb * HEADS + h) * DH + dh)) * SEQ + ss
                    : ((size_t)((bb * HEADS + h) * SEQ + ss)) * DH + dh;
                out[idx] = f2bf(v);
            }
        }
    }
}

// ---------------------------------------------------------------------------
// Flash attention. One block = 8 waves = 128 consecutive queries of ONE
// (b,h). Per 32-key chunk, the block stages K (12KB, contiguous) and V
// (12KB, transposed) into LDS once via global_load_async_to_lds_b128; the 8
// waves then feed their WMMAs from LDS (8x global-traffic reduction).
// Energy is computed TRANSPOSED (K @ Q^T) so the D-tile layout coincides
// with the A-operand layout needed by P @ V — no cross-lane transposes.
// ---------------------------------------------------------------------------
__global__ __launch_bounds__(256)
void attn_kernel(const unsigned short* __restrict__ Qb,
                 const unsigned short* __restrict__ Kb,
                 const unsigned short* __restrict__ Vt,
                 unsigned short* __restrict__ attn)
{
    __shared__ __align__(16) unsigned short Ks[32 * DH];   // [key][dh], 12KB
    __shared__ __align__(16) unsigned short Vs[DH * 32];   // [dh][key], 12KB

    const int tid  = threadIdx.x;
    const int wid  = tid >> 5;                 // wave in block: 0..7
    const int lane = tid & 31;
    const int r = lane & 15, hi = lane >> 4;

    const int bh   = blockIdx.x >> 4;          // 32 (b,h) pairs
    const int qblk = blockIdx.x & 15;          // 16 blocks of 128 queries
    const int q0   = qblk * 128 + wid * 16;

    const unsigned short* Qh = Qb + (size_t)bh * SEQ * DH;
    const unsigned short* Kh = Kb + (size_t)bh * SEQ * DH;
    const unsigned short* Vh = Vt + (size_t)bh * DH * SEQ;

    const unsigned ks_base = (unsigned)(size_t)&Ks[0];
    const unsigned vs_base = (unsigned)(size_t)&Vs[0];

    // Q resident as B-operand (lane = query column r): 48 VGPRs.
    Tile qt[6];
    #pragma unroll
    for (int d = 0; d < 6; ++d)
        qt[d] = load_bf16_tile(Qh + (size_t)(q0 + r) * DH + d * 32 + hi * 8);

    const float scale = 0.07216878f;           // 1/sqrt(192)
    float m_run = -1e30f, l_run = 0.0f;
    v8f oacc[12] = {};                         // 16 queries x 192 dh, 96 VGPRs

    for (int k0 = 0; k0 < SEQ; k0 += 32) {
        __syncthreads();                       // previous chunk fully consumed

        // Stage K chunk: rows k0..k0+31 are contiguous (32*192 bf16 = 12KB).
        const char* ksrc = (const char*)(Kh + (size_t)k0 * DH);
        #pragma unroll
        for (int i = 0; i < 6; ++i) {
            const unsigned off = (unsigned)tid * 16u + (unsigned)i * 4096u;
            async_copy16(ks_base + off, ksrc + off);
        }
        // Stage V chunk: 192 rows of 64B from Vt (stride SEQ), [dh][32] in LDS.
        #pragma unroll
        for (int i = 0; i < 3; ++i) {
            const int idx = tid + i * 256;     // 768 x 16B chunks
            const int seg = idx >> 2, part = idx & 3;
            async_copy16(vs_base + (unsigned)(seg * 64 + part * 16),
                         Vh + (size_t)seg * SEQ + k0 + part * 8);
        }
        wait_async();
        __syncthreads();

        // energy^T: two 16-key x 16-query tiles, K-reduced over Dh=192.
        v8f e0 = {}, e1 = {};
        #pragma unroll
        for (int d = 0; d < 6; ++d) {
            Tile ka = load_bf16_tile(&Ks[(     r) * DH + d * 32 + hi * 8]);
            Tile kb = load_bf16_tile(&Ks[(16 + r) * DH + d * 32 + hi * 8]);
            e0 = wmma_bf16(ka, qt[d], e0);
            e1 = wmma_bf16(kb, qt[d], e1);
        }

        // Online softmax for query q = r (each lane owns 16 of the 32 keys).
        float vals[16];
        #pragma unroll
        for (int j = 0; j < 8; ++j) { vals[j] = e0[j] * scale; vals[8 + j] = e1[j] * scale; }
        float lmax = vals[0];
        #pragma unroll
        for (int j = 1; j < 16; ++j) lmax = fmaxf(lmax, vals[j]);
        lmax = fmaxf(lmax, __shfl_xor(lmax, 16, 32));
        const float m_new = fmaxf(m_run, lmax);
        const float alpha = __expf(m_run - m_new);

        Tile P;                                // lands exactly in A-operand layout
        float lsum = 0.0f;
        #pragma unroll
        for (int j = 0; j < 16; ++j) {
            const float p = __expf(vals[j] - m_new);
            lsum += p;
            P.s[j] = f2bf(p);
        }
        lsum += __shfl_xor(lsum, 16, 32);
        l_run = l_run * alpha + lsum;
        m_run = m_new;

        // Broadcast alpha into C-layout (row m = e + 8*hi) and rescale O.
        float af[8];
        #pragma unroll
        for (int e = 0; e < 8; ++e) af[e] = __shfl(alpha, e + 8 * hi, 32);
        #pragma unroll
        for (int t = 0; t < 12; ++t)
            #pragma unroll
            for (int e = 0; e < 8; ++e) oacc[t][e] *= af[e];

        // O += P(16x32) @ Vchunk(32x192); Vs rows contiguous in keys.
        #pragma unroll
        for (int t = 0; t < 12; ++t) {
            Tile v = load_bf16_tile(&Vs[(t * 16 + r) * 32 + hi * 8]);
            oacc[t] = wmma_bf16(P, v, oacc[t]);
        }
    }

    // Normalize by l and write bf16 [B,S,EMB] (heads concatenated).
    float rl[8];
    {
        const float inv = 1.0f / l_run;
        #pragma unroll
        for (int e = 0; e < 8; ++e) rl[e] = __shfl(inv, e + 8 * hi, 32);
    }
    const int b = bh / HEADS, h = bh % HEADS;
    #pragma unroll
    for (int t = 0; t < 12; ++t) {
        const int col = h * DH + t * 16 + r;
        #pragma unroll
        for (int e = 0; e < 8; ++e) {
            const int m = q0 + e + 8 * hi;
            attn[((size_t)(b * SEQ + m)) * EMB + col] = f2bf(oacc[t][e] * rl[e]);
        }
    }
}

// ---------------------------------------------------------------------------
// Output GEMM: out = attn(bf16) @ Wo^T(bf16) + bo, fp32 result. 32x64/wave.
// ---------------------------------------------------------------------------
__global__ __launch_bounds__(256)
void out_gemm_kernel(const unsigned short* __restrict__ A,
                     const unsigned short* __restrict__ Wb,
                     const float* __restrict__ bias, float* __restrict__ out)
{
    const int wave = (blockIdx.x * blockDim.x + threadIdx.x) >> 5;
    const int lane = threadIdx.x & 31;
    const int r = lane & 15, hi = lane >> 4;
    const int NT64 = EMB / 64;
    const int tn64 = wave % NT64;
    const int tm   = wave / NT64;
    const int row0 = tm * 32;
    const int n0   = tn64 * 64;

    v8f acc[2][4] = {};
    for (int k0 = 0; k0 < EMB; k0 += 32) {
        Tile a0 = load_bf16_tile(A + (size_t)(row0 +      r) * EMB + k0 + hi * 8);
        Tile a1 = load_bf16_tile(A + (size_t)(row0 + 16 + r) * EMB + k0 + hi * 8);
        #pragma unroll
        for (int t = 0; t < 4; ++t) {
            Tile b = load_bf16_tile(Wb + (size_t)(n0 + t * 16 + r) * EMB + k0 + hi * 8);
            acc[0][t] = wmma_bf16(a0, b, acc[0][t]);
            acc[1][t] = wmma_bf16(a1, b, acc[1][t]);
        }
    }
    #pragma unroll
    for (int ms = 0; ms < 2; ++ms) {
        #pragma unroll
        for (int t = 0; t < 4; ++t) {
            const int n = n0 + t * 16 + r;
            const float bval = bias[n];
            #pragma unroll
            for (int e = 0; e < 8; ++e) {
                const int m = row0 + ms * 16 + e + 8 * hi;
                out[(size_t)m * EMB + n] = acc[ms][t][e] + bval;
            }
        }
    }
}

extern "C" void kernel_launch(void* const* d_in, const int* in_sizes, int n_in,
                              void* d_out, int out_size, void* d_ws, size_t ws_size,
                              hipStream_t stream) {
    const float* q  = (const float*)d_in[0];
    const float* k  = (const float*)d_in[1];
    const float* v  = (const float*)d_in[2];
    const float* Wq = (const float*)d_in[3];
    const float* bq = (const float*)d_in[4];
    const float* Wk = (const float*)d_in[5];
    const float* bk = (const float*)d_in[6];
    const float* Wv = (const float*)d_in[7];
    const float* bv = (const float*)d_in[8];
    const float* Wo = (const float*)d_in[9];
    const float* bo = (const float*)d_in[10];

    const size_t act_n = (size_t)BATCH * SEQ * EMB;   // 12.58M elems
    const size_t w_n   = (size_t)EMB * EMB;           // 589824 elems

    unsigned short* p = (unsigned short*)d_ws;
    unsigned short* qx  = p; p += act_n;   // bf16 activations
    unsigned short* kx  = p; p += act_n;
    unsigned short* vx  = p; p += act_n;
    unsigned short* Wqb = p; p += w_n;     // bf16 weights
    unsigned short* Wkb = p; p += w_n;
    unsigned short* Wvb = p; p += w_n;
    unsigned short* Wob = p; p += w_n;
    unsigned short* Qbf = p; p += act_n;   // projected Q/K, [B,H,S,Dh]
    unsigned short* Kbf = p; p += act_n;
    unsigned short* Vt  = p; p += act_n;   // projected V, [B,H,Dh,S]
    unsigned short* attn= p; p += act_n;   // attention out, [B,S,EMB]

    const int cvt_act_blocks = (int)(act_n / 4 / 256);  // 12288
    const int cvt_w_blocks   = (int)(w_n   / 4 / 256);  // 576
    cvt_bf16_kernel<<<cvt_act_blocks, 256, 0, stream>>>(q,  qx);
    cvt_bf16_kernel<<<cvt_act_blocks, 256, 0, stream>>>(k,  kx);
    cvt_bf16_kernel<<<cvt_act_blocks, 256, 0, stream>>>(v,  vx);
    cvt_bf16_kernel<<<cvt_w_blocks,   256, 0, stream>>>(Wq, Wqb);
    cvt_bf16_kernel<<<cvt_w_blocks,   256, 0, stream>>>(Wk, Wkb);
    cvt_bf16_kernel<<<cvt_w_blocks,   256, 0, stream>>>(Wv, Wvb);
    cvt_bf16_kernel<<<cvt_w_blocks,   256, 0, stream>>>(Wo, Wob);

    const int gemm_blocks = (BATCH * SEQ / 32) * (EMB / 64) / 8;   // 768
    const int attn_blocks = (BATCH * HEADS) * (SEQ / 128);         // 512

    proj_kernel<<<gemm_blocks, 256, 0, stream>>>(qx, Wqb, bq, Qbf, 0);
    proj_kernel<<<gemm_blocks, 256, 0, stream>>>(kx, Wkb, bk, Kbf, 0);
    proj_kernel<<<gemm_blocks, 256, 0, stream>>>(vx, Wvb, bv, Vt, 1);
    attn_kernel<<<attn_blocks, 256, 0, stream>>>(Qbf, Kbf, Vt, attn);
    out_gemm_kernel<<<gemm_blocks, 256, 0, stream>>>(attn, Wob, bo, (float*)d_out);
}
